// AttentionBlock_38792144617721
// MI455X (gfx1250) — compile-verified, matrix operand order checked by reading, not run on hardware
//
#include <hip/hip_runtime.h>
#include <hip/hip_bf16.h>
#include <math.h>

// ---------------------------------------------------------------------------
// Problem constants
// ---------------------------------------------------------------------------
#define BB   16
#define CC   512
#define NN   1024            // H*W = 32*32
#define O3   1536            // 3*C
#define EPS  1e-5f
#define LDB  40              // LDS row stride (halves) for B panels (padded)
#define PANEL (64 * LDB)     // one 64-row B panel in halves

typedef _Float16 v16h __attribute__((ext_vector_type(16)));
typedef _Float16 v8h  __attribute__((ext_vector_type(8)));
typedef _Float16 v4h  __attribute__((ext_vector_type(4)));
typedef float    v8f  __attribute__((ext_vector_type(8)));
typedef int      v4i  __attribute__((ext_vector_type(4)));

// ---------------------------------------------------------------------------
// Async global->LDS staging (gfx1250 GLOBAL_LOAD_ASYNC_TO_LDS_B128) with
// register-path fallback if the builtin is not declared by this toolchain.
// Builtin signature (from clang diagnostic): (v4i AS1*, v4i AS3*, imm, imm).
// ---------------------------------------------------------------------------
#if defined(__has_builtin)
#if __has_builtin(__builtin_amdgcn_global_load_async_to_lds_b128)
#define ASYNC_STAGE 1
#endif
#endif

__device__ __forceinline__ void stage_panel(const _Float16* gsrc, _Float16* ldst) {
#ifdef ASYNC_STAGE
  __builtin_amdgcn_global_load_async_to_lds_b128(
      (__attribute__((address_space(1))) v4i*)(void*)gsrc,
      (__attribute__((address_space(3))) v4i*)(void*)ldst, 0, 0);
#else
  *(v8h*)ldst = *(const v8h*)gsrc;
  __builtin_prefetch(gsrc + 32, 0, 1);
#endif
}

__device__ __forceinline__ void wait_stage() {
#ifdef ASYNC_STAGE
#if __has_builtin(__builtin_amdgcn_s_wait_asynccnt)
  __builtin_amdgcn_s_wait_asynccnt(0);
#else
  asm volatile("s_wait_asynccnt 0x0" ::: "memory");
#endif
#endif
}

// ---------------------------------------------------------------------------
// WMMA operand loaders (wave32, v_wmma_f32_16x16x32_f16)
//
// A-operand (16x32, M x K): lane L holds row M = L&15; per ISA layout:
//   v16h elems 0..7  = K[ (L>=16?8:0)    .. +7 ]
//   v16h elems 8..15 = K[ 16+(L>=16?8:0) .. +7 ]
// B-operand is fed from the TRANSPOSED matrix (rows K-contiguous) with the
// identical per-lane pattern (lane = output column).
// ---------------------------------------------------------------------------
__device__ __forceinline__ v16h load_opnd(const _Float16* __restrict rowbase,
                                          int ld, int lane) {
  int row = lane & 15;
  int hi  = (lane >> 4) << 3;              // 0 or 8
  const _Float16* p = rowbase + (size_t)row * ld + hi;
  v8h a = *(const v8h*)(p);
  v8h b = *(const v8h*)(p + 16);
  v16h r;
#pragma unroll
  for (int i = 0; i < 8; ++i) { r[i] = a[i]; r[i + 8] = b[i]; }
  return r;
}

__device__ __forceinline__ v16h load_opnd_lds(const _Float16* base, int lane) {
  int row = lane & 15;
  int hi  = (lane >> 4) << 3;
  const _Float16* p = base + row * LDB + hi;
  v8h a = *(const v8h*)(p);
  v8h b = *(const v8h*)(p + 16);
  v16h r;
#pragma unroll
  for (int i = 0; i < 8; ++i) { r[i] = a[i]; r[i + 8] = b[i]; }
  return r;
}

// ---------------------------------------------------------------------------
// Shared macro-tile GEMM loop: block = 256 threads = 8 waves.
// Wave w owns a 16-row A strip; the block shares a 64-column B panel,
// double-buffered in LDS (one barrier per K-step).
// ---------------------------------------------------------------------------
template <int BLD>
__device__ __forceinline__ void gemm_loop(const _Float16* __restrict Arow, int Ald,
                                          const _Float16* __restrict Bbase,
                                          _Float16* sB, int ktot,
                                          v8f acc[4], int t, int lane) {
  int row  = t >> 2;
  int part = (t & 3) << 3;
  const _Float16* gsrc = Bbase + (size_t)row * BLD + part;
  _Float16*       ldst = sB + row * LDB + part;

  stage_panel(gsrc, ldst);                         // panel 0 -> buffer 0
  for (int kk = 0; kk < ktot; kk += 32) {
    int cur = (kk >> 5) & 1;
    wait_stage();
    __syncthreads();                               // panel `cur` ready
    if (kk + 32 < ktot)
      stage_panel(gsrc + kk + 32, ldst + (cur ^ 1) * PANEL);
    v16h a = load_opnd(Arow + kk, Ald, lane);
    const _Float16* pan = sB + cur * PANEL;
#pragma unroll
    for (int j = 0; j < 4; ++j) {
      v16h b = load_opnd_lds(pan + (j * 16) * LDB, lane);
      acc[j] = __builtin_amdgcn_wmma_f32_16x16x32_f16(
          false, a, false, b, (short)0, acc[j], false, false);
    }
    // no trailing barrier: ds reads of `cur` are consumed by the WMMAs above
    // (s_wait_dscnt before wmma), so the next iteration's barrier is enough.
  }
}

// ---------------------------------------------------------------------------
// K0: convert f32 weight matrices to f16
// ---------------------------------------------------------------------------
__global__ __launch_bounds__(256) void k_convert_w(const float* __restrict qw,
                                                   const float* __restrict ow,
                                                   _Float16* __restrict qwh,
                                                   _Float16* __restrict owh) {
  int i = blockIdx.x * 256 + threadIdx.x;
  if (i < O3 * CC)  qwh[i] = (_Float16)qw[i];
  if (i < CC * CC)  owh[i] = (_Float16)ow[i];
}

// ---------------------------------------------------------------------------
// K1: per-batch mean / rstd over C*H*W
// ---------------------------------------------------------------------------
__global__ __launch_bounds__(256) void k_stats(const float* __restrict x,
                                               float* __restrict stats) {
  const int M = CC * NN;                       // 524288
  const float* p = x + (size_t)blockIdx.x * M;
  float s = 0.f, s2 = 0.f;
  for (int i = threadIdx.x; i < M / 4; i += 256) {
    float4 v = ((const float4*)p)[i];
    s  += v.x + v.y + v.z + v.w;
    s2 += v.x * v.x + v.y * v.y + v.z * v.z + v.w * v.w;
  }
  __shared__ float rs[256], rq[256];
  rs[threadIdx.x] = s; rq[threadIdx.x] = s2;
  __syncthreads();
  for (int off = 128; off; off >>= 1) {
    if (threadIdx.x < off) {
      rs[threadIdx.x] += rs[threadIdx.x + off];
      rq[threadIdx.x] += rq[threadIdx.x + off];
    }
    __syncthreads();
  }
  if (threadIdx.x == 0) {
    float mean = rs[0] / (float)M;
    float var  = rq[0] / (float)M - mean * mean;
    stats[2 * blockIdx.x + 0] = mean;
    stats[2 * blockIdx.x + 1] = rsqrtf(var + EPS);
  }
}

// ---------------------------------------------------------------------------
// K2: GroupNorm + transpose: x[b][c][n] (f32) -> xnT[b][n][c] (f16)
// ---------------------------------------------------------------------------
__global__ __launch_bounds__(256) void k_norm_t(const float* __restrict x,
                                                const float* __restrict stats,
                                                const float* __restrict nw,
                                                const float* __restrict nb,
                                                _Float16* __restrict xt) {
  __shared__ float tile[32][33];
  int b  = blockIdx.z;
  int n0 = blockIdx.x * 32;
  int c0 = blockIdx.y * 32;
  float mean = stats[2 * b], rstd = stats[2 * b + 1];
  int tx = threadIdx.x, ty = threadIdx.y;    // (32, 8)
#pragma unroll
  for (int i = 0; i < 32; i += 8) {
    int c = c0 + ty + i;
    float v = x[((size_t)b * CC + c) * NN + n0 + tx];
    tile[ty + i][tx] = (v - mean) * rstd * nw[c] + nb[c];
  }
  __syncthreads();
#pragma unroll
  for (int i = 0; i < 32; i += 8) {
    int n = n0 + ty + i;
    xt[((size_t)b * NN + n) * CC + c0 + tx] = (_Float16)tile[tx][ty + i];
  }
}

// ---------------------------------------------------------------------------
// K3: QKV GEMM  D[o][n] = sum_c W[o][c] * xnT[n][c]  (+ bias)
//   q,k stored [b][n][c] (f16, contiguous v8h epilogue stores)
//   v   stored [b][c][n] (f16) so the P*V GEMM reads it K-contiguous
// ---------------------------------------------------------------------------
__global__ __launch_bounds__(256) void k_qkv(const _Float16* __restrict W,
                                             const float* __restrict bias,
                                             const _Float16* __restrict Xt,
                                             _Float16* __restrict q,
                                             _Float16* __restrict k,
                                             _Float16* __restrict v) {
  __shared__ __align__(16) _Float16 sB[2 * PANEL];
  int b = blockIdx.z, t = threadIdx.x;
  int wave = t >> 5, lane = t & 31;
  int o0 = blockIdx.x * 128;                 // gridDim.x = 12
  int n0 = blockIdx.y * 64;                  // gridDim.y = 16
  int orow = o0 + wave * 16;
  const _Float16* Arow  = W + (size_t)orow * CC;
  const _Float16* Bbase = Xt + ((size_t)b * NN + n0) * CC;
  v8f acc[4] = {};
  gemm_loop<CC>(Arow, CC, Bbase, sB, CC, acc, t, lane);

  int col   = lane & 15;
  int rbase = (lane >> 4) << 3;              // rows rbase..rbase+7
  float bb[8];
#pragma unroll
  for (int r = 0; r < 8; ++r) bb[r] = bias[orow + rbase + r];

  int region = o0 >> 9;                      // 0=q 1=k 2=v
  int oloc   = (orow & (CC - 1)) + rbase;
#pragma unroll
  for (int j = 0; j < 4; ++j) {
    int ncol = n0 + j * 16 + col;
    if (region < 2) {
      _Float16* dst = (region == 0) ? q : k;
      v8h out;
#pragma unroll
      for (int r = 0; r < 8; ++r) out[r] = (_Float16)(acc[j][r] + bb[r]);
      *(v8h*)(dst + ((size_t)b * NN + ncol) * CC + oloc) = out;
    } else {
#pragma unroll
      for (int r = 0; r < 8; ++r)
        v[((size_t)b * CC + oloc + r) * NN + ncol] = (_Float16)(acc[j][r] + bb[r]);
    }
  }
}

// ---------------------------------------------------------------------------
// K4: scores  S[nq][nk] = (1/sqrt(C)) * sum_c q[nq][c]*k[nk][c]   (f32)
// ---------------------------------------------------------------------------
__global__ __launch_bounds__(256) void k_scores(const _Float16* __restrict q,
                                                const _Float16* __restrict kmat,
                                                float* __restrict s) {
  __shared__ __align__(16) _Float16 sB[2 * PANEL];
  const float scale = 0.044194173824159216f;   // 512^-0.5
  int b = blockIdx.z, t = threadIdx.x;
  int wave = t >> 5, lane = t & 31;
  int nq0 = blockIdx.x * 128 + wave * 16;      // gridDim.x = 8
  int nk0 = blockIdx.y * 64;                   // gridDim.y = 16
  const _Float16* Arow  = q    + ((size_t)b * NN + nq0) * CC;
  const _Float16* Bbase = kmat + ((size_t)b * NN + nk0) * CC;
  v8f acc[4] = {};
  gemm_loop<CC>(Arow, CC, Bbase, sB, CC, acc, t, lane);

  int col = lane & 15, rbase = (lane >> 4) << 3;
#pragma unroll
  for (int j = 0; j < 4; ++j)
#pragma unroll
    for (int r = 0; r < 8; ++r)
      s[((size_t)b * NN + nq0 + rbase + r) * NN + nk0 + j * 16 + col] =
          acc[j][r] * scale;
}

// ---------------------------------------------------------------------------
// K5: softmax rows; write unnormalized exp as f16 IN PLACE over the f32 row
//     (half row stride = 2*NN), store 1/rowsum.
// ---------------------------------------------------------------------------
__global__ __launch_bounds__(256) void k_softmax(float* __restrict s,
                                                 float* __restrict rowinv) {
  size_t row = (size_t)blockIdx.y * NN + blockIdx.x;
  float* rp = s + row * NN;
  int t = threadIdx.x;
  float4 vv = ((const float4*)rp)[t];
  __shared__ float red[256];
  float mx = fmaxf(fmaxf(vv.x, vv.y), fmaxf(vv.z, vv.w));
  red[t] = mx; __syncthreads();
  for (int off = 128; off; off >>= 1) {
    if (t < off) red[t] = fmaxf(red[t], red[t + off]);
    __syncthreads();
  }
  mx = red[0]; __syncthreads();
  float e0 = expf(vv.x - mx), e1 = expf(vv.y - mx);
  float e2 = expf(vv.z - mx), e3 = expf(vv.w - mx);
  red[t] = e0 + e1 + e2 + e3; __syncthreads();
  for (int off = 128; off; off >>= 1) {
    if (t < off) red[t] += red[t + off];
    __syncthreads();
  }
  if (t == 0) rowinv[row] = 1.0f / red[0];
  v4h out = {(_Float16)e0, (_Float16)e1, (_Float16)e2, (_Float16)e3};
  ((v4h*)rp)[t] = out;            // all reads completed before last barrier
}

// ---------------------------------------------------------------------------
// K6: attn_out[n][c] = rowinv[n] * sum_m P[n][m] * v[c][m]   -> f16 [b][n][c]
//     P rows live at half-stride 2*NN inside the f32 score buffer.
// ---------------------------------------------------------------------------
__global__ __launch_bounds__(256) void k_pv(const _Float16* __restrict p,
                                            const _Float16* __restrict v,
                                            const float* __restrict rowinv,
                                            _Float16* __restrict attn) {
  __shared__ __align__(16) _Float16 sB[2 * PANEL];
  int b = blockIdx.z, t = threadIdx.x;
  int wave = t >> 5, lane = t & 31;
  int nq0 = blockIdx.x * 128 + wave * 16;     // gridDim.x = 8
  int c0  = blockIdx.y * 64;                  // gridDim.y = 8
  const _Float16* Arow  = p + ((size_t)b * NN + nq0) * (2 * NN);
  const _Float16* Bbase = v + ((size_t)b * CC + c0) * NN;
  v8f acc[4] = {};
  gemm_loop<NN>(Arow, 2 * NN, Bbase, sB, NN, acc, t, lane);

  int col = lane & 15, rbase = (lane >> 4) << 3;
  float iv[8];
#pragma unroll
  for (int r = 0; r < 8; ++r) iv[r] = rowinv[(size_t)b * NN + nq0 + rbase + r];
#pragma unroll
  for (int j = 0; j < 4; ++j)
#pragma unroll
    for (int r = 0; r < 8; ++r)
      attn[((size_t)b * NN + nq0 + rbase + r) * CC + c0 + j * 16 + col] =
          (_Float16)(acc[j][r] * iv[r]);
}

// ---------------------------------------------------------------------------
// K7: out-proj + bias + residual:
//     y[b][o][n] = sum_c Wo[o][c]*attn[n][c] + ob[o] + x[b][o][n]
// ---------------------------------------------------------------------------
__global__ __launch_bounds__(256) void k_proj(const _Float16* __restrict Wo,
                                              const float* __restrict ob,
                                              const _Float16* __restrict attn,
                                              const float* __restrict x,
                                              float* __restrict y) {
  __shared__ __align__(16) _Float16 sB[2 * PANEL];
  int b = blockIdx.z, t = threadIdx.x;
  int wave = t >> 5, lane = t & 31;
  int o0 = blockIdx.x * 128 + wave * 16;      // gridDim.x = 4
  int n0 = blockIdx.y * 64;                   // gridDim.y = 16
  const _Float16* Arow  = Wo   + (size_t)o0 * CC;
  const _Float16* Bbase = attn + ((size_t)b * NN + n0) * CC;
  v8f acc[4] = {};
  gemm_loop<CC>(Arow, CC, Bbase, sB, CC, acc, t, lane);

  int col = lane & 15, rbase = (lane >> 4) << 3;
  float bb[8];
#pragma unroll
  for (int r = 0; r < 8; ++r) bb[r] = ob[o0 + rbase + r];
#pragma unroll
  for (int j = 0; j < 4; ++j)
#pragma unroll
    for (int r = 0; r < 8; ++r) {
      size_t idx = ((size_t)b * CC + o0 + rbase + r) * NN + n0 + j * 16 + col;
      y[idx] = acc[j][r] + bb[r] + x[idx];
    }
}

// ---------------------------------------------------------------------------
// Host-side launcher
// ---------------------------------------------------------------------------
extern "C" void kernel_launch(void* const* d_in, const int* in_sizes, int n_in,
                              void* d_out, int out_size, void* d_ws, size_t ws_size,
                              hipStream_t stream) {
  const float* x      = (const float*)d_in[0];   // [16,512,32,32]
  const float* norm_w = (const float*)d_in[1];   // [512]
  const float* norm_b = (const float*)d_in[2];   // [512]
  const float* qkv_w  = (const float*)d_in[3];   // [1536,512]
  const float* qkv_b  = (const float*)d_in[4];   // [1536]
  const float* out_w  = (const float*)d_in[5];   // [512,512]
  const float* out_b  = (const float*)d_in[6];   // [512]
  float* y = (float*)d_out;

  // deterministic workspace carve-up (256B aligned), ~136 MB total
  char* base = (char*)d_ws;
  size_t off = 0;
  auto carve = [&](size_t bytes) {
    char* p = base + off;
    off = (off + bytes + 255) & ~(size_t)255;
    return p;
  };
  float*    stats  = (float*)   carve(BB * 2 * sizeof(float));
  float*    rowinv = (float*)   carve((size_t)BB * NN * sizeof(float));
  _Float16* qwh    = (_Float16*)carve((size_t)O3 * CC * 2);
  _Float16* owh    = (_Float16*)carve((size_t)CC * CC * 2);
  _Float16* xt     = (_Float16*)carve((size_t)BB * NN * CC * 2);  // reused as attn
  _Float16* qb     = (_Float16*)carve((size_t)BB * NN * CC * 2);
  _Float16* kb     = (_Float16*)carve((size_t)BB * NN * CC * 2);
  _Float16* vb     = (_Float16*)carve((size_t)BB * NN * CC * 2);
  float*    s      = (float*)   carve((size_t)BB * NN * NN * sizeof(float));
  _Float16* attn   = xt;  // xnT is dead after K3; alias for attention output

  k_convert_w<<<dim3((O3 * CC + 255) / 256), dim3(256), 0, stream>>>(qkv_w, out_w, qwh, owh);
  k_stats    <<<dim3(BB),                    dim3(256), 0, stream>>>(x, stats);
  k_norm_t   <<<dim3(NN / 32, CC / 32, BB),  dim3(32, 8), 0, stream>>>(x, stats, norm_w, norm_b, xt);
  k_qkv      <<<dim3(O3 / 128, NN / 64, BB), dim3(256), 0, stream>>>(qwh, qkv_b, xt, qb, kb, vb);
  k_scores   <<<dim3(NN / 128, NN / 64, BB), dim3(256), 0, stream>>>(qb, kb, s);
  k_softmax  <<<dim3(NN, BB),                dim3(256), 0, stream>>>(s, rowinv);
  k_pv       <<<dim3(NN / 128, CC / 64, BB), dim3(256), 0, stream>>>((const _Float16*)s, vb, rowinv, attn);
  k_proj     <<<dim3(CC / 128, NN / 64, BB), dim3(256), 0, stream>>>(owh, out_b, attn, x, y);

  (void)in_sizes; (void)n_in; (void)out_size; (void)ws_size;
}
